// lstm_decoder_16853451670106
// MI455X (gfx1250) — matrix operand order repaired and checked by reading, not compiled, not run on hardware
//
#include <hip/hip_runtime.h>

// LSTM decoder: B=32768, LATENT=128, H=17, 4H=68, T=50.
// f16 WMMA (f32 accumulate) for the latent projection GEMM and the per-step
// recurrent GEMM; fp32 VALU for gate math. TDM (tensor_load_to_lds) stages the
// per-wave x tile global->LDS. One wave owns 16 samples; 8 waves/block.

typedef __attribute__((ext_vector_type(16))) _Float16 v16h;
typedef __attribute__((ext_vector_type(8)))  _Float16 v8h;
typedef __attribute__((ext_vector_type(8)))  float    v8f;
typedef __attribute__((ext_vector_type(4)))  unsigned v4u;
typedef __attribute__((ext_vector_type(8)))  int      v8i;
typedef __attribute__((ext_vector_type(4)))  int      v4i;

#define NSAMPLES   32768
#define LATENT     128
#define HID        17
#define GATES      68
#define STEPS      50
#define NT1        7            // phase-1 N tiles (112 cols)
#define NGT        5            // gate N tiles (80 cols)
#define WAVES      8
#define BLK_SAMPLES 128
#define XF_BYTES   (WAVES * 16 * LATENT * 4)   // 65536: f32 x staging (aliases sh_buf)

__device__ inline float fsigm(float v) {
    return __builtin_amdgcn_rcpf(1.0f + __expf(-v));
}
__device__ inline float ftanh(float v) {
    return 1.0f - 2.0f * __builtin_amdgcn_rcpf(__expf(2.0f * v) + 1.0f);
}

typedef __attribute__((address_space(3))) void lds_void;
__device__ inline unsigned lds_offset_of(void* p) {
    return (unsigned)(unsigned long long)(lds_void*)p;
}

// Build a 16x32 f16 A-fragment from an f32 row in LDS (K pairs are contiguous).
__device__ inline v16h buildA_f32(const float* rowp, int kbase, int hs) {
    v16h a;
#pragma unroll
    for (int p = 0; p < 4; ++p) {
        float2 f = *(const float2*)(rowp + kbase + hs * 16 + 2 * p);
        a[2 * p]     = (_Float16)f.x;
        a[2 * p + 1] = (_Float16)f.y;
        float2 g = *(const float2*)(rowp + kbase + hs * 16 + 8 + 2 * p);
        a[8 + 2 * p]     = (_Float16)g.x;
        a[8 + 2 * p + 1] = (_Float16)g.y;
    }
    return a;
}

__device__ inline v16h combine16(v8h lo, v8h hi) {
    v16h a;
#pragma unroll
    for (int i = 0; i < 8; ++i) { a[i] = lo[i]; a[i + 8] = hi[i]; }
    return a;
}

__global__ __launch_bounds__(256) void lstm_decoder_kernel(
    const float* __restrict__ x,    const float* __restrict__ Wh0,
    const float* __restrict__ bh0,  const float* __restrict__ Wc0,
    const float* __restrict__ bc0,  const float* __restrict__ Wih,
    const float* __restrict__ Whh,  const float* __restrict__ bih,
    const float* __restrict__ bhh,  float* __restrict__ out)
{
    // [0, 64K):  f32 x staging [8][16][128]  (phase 1) -- later reused as
    //            f32 result/gate staging [8][16][112]
    // [64K,72K): f16 h state [8][16][32] (K padded to 32 with zeros)
    __shared__ __align__(16) char smem[XF_BYTES + WAVES * 16 * 32 * 2];
    __shared__ float sh_bias[112];

    float*    sh_xf  = (float*)smem;
    float*    sh_buf = (float*)smem;                  // aliases sh_xf (see barrier)
    _Float16* sh_h   = (_Float16*)(smem + XF_BYTES);

    const int tid  = threadIdx.x;
    const int wave = tid >> 5;
    const int lane = tid & 31;
    const int nl   = lane & 15;
    const int hs   = lane >> 4;
    const int s0   = blockIdx.x * BLK_SAMPLES + wave * 16;

    // ---- stage x tile [16,128] f32 global -> LDS ----
#if __has_builtin(__builtin_amdgcn_tensor_load_to_lds)
    {
        // Tensor DMA descriptor (ISA 8.3/8.4): 2-D tensor [32768,128] f32,
        // tile [16,128] starting at this wave's first sample.
        unsigned long long ga =
            (unsigned long long)(const void*)(x + (size_t)s0 * LATENT);
        unsigned ldsa = lds_offset_of(smem) + (unsigned)(wave * 16 * LATENT * 4);
        v4u g0;
        g0[0] = 1u;                                    // count=1, user descriptor
        g0[1] = ldsa;                                  // lds_addr (bytes)
        g0[2] = (unsigned)ga;                          // global_addr[31:0]
        g0[3] = (unsigned)((ga >> 32) & 0x01FFFFFFull) // global_addr[56:32]
                | (2u << 30);                          // type=2 ("image")
        v8i g1;
        g1[0] = (int)(2u << 16);                       // data_size=4B; wg_mask=0
        g1[1] = (int)(128u << 16);                     // tensor_dim0 = 128
        g1[2] = (int)(32768u << 16);                   // tensor_dim1 = 32768 (lo16)
        g1[3] = (int)(128u << 16);                     // tile_dim0 = 128
        g1[4] = 16;                                    // tile_dim1 = 16
        g1[5] = 128;                                   // tensor_dim0_stride = 128
        g1[6] = (int)(128u << 16);                     // tensor_dim1_stride = 128
        g1[7] = 0;
        v4i gz = (v4i){0, 0, 0, 0};
#if __has_include(<hip/amd_detail/amd_gfx1250_TDM.h>)
        v8i gz8 = (v8i){0, 0, 0, 0, 0, 0, 0, 0};
        __builtin_amdgcn_tensor_load_to_lds(g0, g1, gz, gz, gz8, 0);
#else
        __builtin_amdgcn_tensor_load_to_lds(g0, g1, gz, gz, 0);
#endif
    }
#else
    {
        const int m = lane >> 1, half = lane & 1;
        const float4* src = (const float4*)(x + (size_t)(s0 + m) * LATENT + half * 64);
        float4* dst = (float4*)(sh_xf + (size_t)(wave * 16 + m) * LATENT + half * 64);
#pragma unroll
        for (int v = 0; v < 16; ++v) dst[v] = src[v];
    }
#endif

    // ---- fused bias vector: [b_h0 | b_c0 | b_ih+b_hh | 0-pad] (overlaps DMA) ----
    if (tid < 112) {
        float bv = 0.0f;
        if      (tid < 17)  bv = bh0[tid];
        else if (tid < 34)  bv = bc0[tid - 17];
        else if (tid < 102) bv = bih[tid - 34] + bhh[tid - 34];
        sh_bias[tid] = bv;
    }
    __syncthreads();

#if __has_builtin(__builtin_amdgcn_tensor_load_to_lds)
#if __has_builtin(__builtin_amdgcn_s_wait_tensorcnt)
    __builtin_amdgcn_s_wait_tensorcnt(0);
#else
    asm volatile("s_wait_tensorcnt 0" ::: "memory");
#endif
#endif
    asm volatile("s_wait_dscnt 0" ::: "memory");

    // ================= Phase 1: [16,128] x [128,112] via WMMA =================
    v8f C1[NT1];
#pragma unroll
    for (int t = 0; t < NT1; ++t) {
        float bv = sh_bias[t * 16 + nl];
        C1[t] = (v8f){bv, bv, bv, bv, bv, bv, bv, bv};
    }

    const float* xrow = sh_xf + (size_t)(wave * 16 + nl) * LATENT;
#pragma unroll
    for (int kc = 0; kc < 4; ++kc) {
        v16h A = buildA_f32(xrow, kc * 32, hs);
#pragma unroll
        for (int t = 0; t < NT1; ++t) {
            const int p = t * 16 + nl;             // fused output row
            const float* row = nullptr;
            if      (p < 17)  row = Wh0 + p * LATENT;
            else if (p < 34)  row = Wc0 + (p - 17) * LATENT;
            else if (p < 102) row = Wih + (p - 34) * LATENT;
            v16h Bf;
            if (row) {
                const float4* r4 = (const float4*)(row + kc * 32 + hs * 16);
#pragma unroll
                for (int q = 0; q < 4; ++q) {
                    float4 f = r4[q];
                    Bf[q * 4 + 0] = (_Float16)f.x; Bf[q * 4 + 1] = (_Float16)f.y;
                    Bf[q * 4 + 2] = (_Float16)f.z; Bf[q * 4 + 3] = (_Float16)f.w;
                }
            } else {
#pragma unroll
                for (int j = 0; j < 16; ++j) Bf[j] = (_Float16)0.0f;
            }
            C1[t] = __builtin_amdgcn_wmma_f32_16x16x32_f16(
                        false, A, false, Bf, (short)0, C1[t], false, false);
        }
    }

    // sh_buf aliases sh_xf and the per-wave slices differ -> block barrier
    __syncthreads();

    // ---- stage phase-1 result [16,112] to LDS for re-alignment ----
#pragma unroll
    for (int t = 0; t < NT1; ++t)
#pragma unroll
        for (int r = 0; r < 8; ++r)
            sh_buf[(size_t)(wave * 16 + r + 8 * hs) * 112 + t * 16 + nl] = C1[t][r];
    asm volatile("s_wait_dscnt 0" ::: "memory");

    // xg -> 5 gate-tile accumulator fragments (persist as WMMA C input)
    v8f XG[NGT];
#pragma unroll
    for (int gt = 0; gt < NGT; ++gt) {
        const int gate = gt * 16 + nl;
#pragma unroll
        for (int r = 0; r < 8; ++r)
            XG[gt][r] = (gate < GATES)
                ? sh_buf[(size_t)(wave * 16 + r + 8 * hs) * 112 + 34 + gate] : 0.0f;
    }

    // c0 -> registers; h0 -> f16 LDS (cols 17..31 zero-padded for K=32 WMMA)
    const int ncols = hs ? 8 : 9;
    const int cbase = hs * 9;
    float creg[9];
#pragma unroll
    for (int i = 0; i < 9; ++i)
        creg[i] = (i < ncols)
            ? sh_buf[(size_t)(wave * 16 + nl) * 112 + 17 + cbase + i] : 0.0f;
#pragma unroll
    for (int j = 0; j < 16; ++j) {
        const int col = hs * 16 + j;
        sh_h[(size_t)(wave * 16 + nl) * 32 + col] =
            (col < HID) ? (_Float16)sh_buf[(size_t)(wave * 16 + nl) * 112 + col]
                        : (_Float16)0.0f;
    }

    // W_hh -> 5 persistent B fragments: B[K][N] = Whh[N][K], K padded 17->32
    v16h BG[NGT];
#pragma unroll
    for (int gt = 0; gt < NGT; ++gt) {
#pragma unroll
        for (int j = 0; j < 16; ++j) {
            const int K = hs * 16 + j;
            const int N = gt * 16 + nl;
            BG[gt][j] = (K < HID && N < GATES) ? (_Float16)Whh[N * HID + K]
                                               : (_Float16)0.0f;
        }
    }
    asm volatile("s_wait_dscnt 0" ::: "memory");

    // ================= Recurrence: 50 sequential steps =================
    const _Float16* hrow = sh_h + (size_t)(wave * 16 + nl) * 32;
    for (int t = 0; t < STEPS; ++t) {
        // A fragment of h: exactly two ds_load_b128 per lane
        v8h lo = *(const v8h*)(hrow + hs * 16);
        v8h hi = *(const v8h*)(hrow + hs * 16 + 8);
        v16h A = combine16(lo, hi);

        // g = xg + h @ Whh^T  (C input = persistent xg fragments)
        v8f D[NGT];
#pragma unroll
        for (int gt = 0; gt < NGT; ++gt)
            D[gt] = __builtin_amdgcn_wmma_f32_16x16x32_f16(
                        false, A, false, BG[gt], (short)0, XG[gt], false, false);

        // stage gates [16,80] into LDS (reuse staging buffer, stride 112)
#pragma unroll
        for (int gt = 0; gt < NGT; ++gt)
#pragma unroll
            for (int r = 0; r < 8; ++r)
                sh_buf[(size_t)(wave * 16 + r + 8 * hs) * 112 + gt * 16 + nl] = D[gt][r];
        asm volatile("s_wait_dscnt 0" ::: "memory");

        // elementwise gate math; lane owns (sample nl, cols cbase..cbase+ncols-1)
        const float* grow = sh_buf + (size_t)(wave * 16 + nl) * 112;
#pragma unroll
        for (int i = 0; i < 9; ++i) {
            if (i < ncols) {
                const int col = cbase + i;
                float iv = grow[col];
                float fv = grow[17 + col];
                float gv = grow[34 + col];
                float ov = grow[51 + col];
                float cc = fsigm(fv) * creg[i] + fsigm(iv) * ftanh(gv);
                creg[i] = cc;
                float hv = fsigm(ov) * ftanh(cc);
                out[(size_t)(s0 + nl) * (STEPS * HID) + t * HID + col] = hv;
                sh_h[(size_t)(wave * 16 + nl) * 32 + col] = (_Float16)hv;
            }
        }
        asm volatile("s_wait_dscnt 0" ::: "memory");  // h visible to next A build
    }
}

extern "C" void kernel_launch(void* const* d_in, const int* in_sizes, int n_in,
                              void* d_out, int out_size, void* d_ws, size_t ws_size,
                              hipStream_t stream) {
    (void)in_sizes; (void)n_in; (void)d_ws; (void)ws_size; (void)out_size;
    const float* x   = (const float*)d_in[0];
    const float* Wh0 = (const float*)d_in[1];
    const float* bh0 = (const float*)d_in[2];
    const float* Wc0 = (const float*)d_in[3];
    const float* bc0 = (const float*)d_in[4];
    const float* Wih = (const float*)d_in[5];
    const float* Whh = (const float*)d_in[6];
    const float* bih = (const float*)d_in[7];
    const float* bhh = (const float*)d_in[8];
    float* out = (float*)d_out;

    dim3 grid(NSAMPLES / BLK_SAMPLES);    // 256 blocks
    dim3 block(256);                       // 8 waves of 32
    lstm_decoder_kernel<<<grid, block, 0, stream>>>(
        x, Wh0, bh0, Wc0, bc0, Wih, Whh, bih, bhh, out);
}
